// RecurrentBase_36490042147134
// MI455X (gfx1250) — compile-verified, matrix operand order checked by reading, not run on hardware
//
#include <hip/hip_runtime.h>
#include <hip/hip_bf16.h>
#include <stdint.h>

// Problem constants (from reference)
#define TT 512
#define BB 128
#define DD 256
#define HH 512

typedef __attribute__((ext_vector_type(16))) __bf16 v16bf;
typedef __attribute__((ext_vector_type(8)))  float  v8f;

// ---------------- workspace layout (bytes) ----------------
static constexpr size_t OFF_W0X = 0;                               // [2048,256] bf16
static constexpr size_t OFF_W0H = OFF_W0X + (size_t)2048*256*2;    // [2048,512] bf16
static constexpr size_t OFF_W1X = OFF_W0H + (size_t)2048*512*2;    // [2048,512] bf16
static constexpr size_t OFF_W1H = OFF_W1X + (size_t)2048*512*2;    // [2048,512] bf16
static constexpr size_t OFF_B0  = OFF_W1H + (size_t)2048*512*2;    // [2048] f32
static constexpr size_t OFF_B1  = OFF_B0  + 2048*4;                // [2048] f32
static constexpr size_t OFF_HA  = OFF_B1  + 2048*4;                // [2][128,512] bf16
static constexpr size_t OFF_HB  = OFF_HA  + (size_t)2*BB*HH*2;     // [2][128,512] bf16
static constexpr size_t OFF_CA  = OFF_HB  + (size_t)2*BB*HH*2;     // [128,512] f32
static constexpr size_t OFF_CB  = OFF_CA  + (size_t)BB*HH*4;       // [128,512] f32
static constexpr size_t STATE_BYTES = OFF_CB + (size_t)BB*HH*4 - OFF_HA;

// ---------------- prologue kernels ----------------
__global__ void cvt_f32_to_bf16(const float* __restrict__ src,
                                __bf16* __restrict__ dst, int n) {
  int i = blockIdx.x * blockDim.x + threadIdx.x;
  if (i < n) dst[i] = (__bf16)src[i];
}

__global__ void bias_sum(const float* __restrict__ a, const float* __restrict__ b,
                         float* __restrict__ out, int n) {
  int i = blockIdx.x * blockDim.x + threadIdx.x;
  if (i < n) out[i] = a[i] + b[i];
}

__global__ void zero_u32(uint32_t* __restrict__ p, int n) {
  int i = blockIdx.x * blockDim.x + threadIdx.x;
  if (i < n) p[i] = 0u;
}

// ---------------- helpers ----------------
__device__ __forceinline__ float fast_sigmoid(float v) {
  return 1.0f / (1.0f + __expf(-v));
}
__device__ __forceinline__ float fast_tanh(float v) {
  return 1.0f - 2.0f / (__expf(2.0f * v) + 1.0f);
}

template <bool XF32>
__device__ __forceinline__ v16bf load_a_frag(const float* __restrict__ xf,
                                             const __bf16* __restrict__ xb,
                                             size_t off) {
  if constexpr (XF32) {
    v16bf a;
    const float* p = xf + off;
#pragma unroll
    for (int i = 0; i < 16; ++i) a[i] = (__bf16)p[i];
    return a;
  } else {
    return *(const v16bf*)(xb + off);
  }
}

// One matrix contribution: acc[g][u] += A(rows b0+u*16..) x W[g*H+j0..]^T
// Single A base + single W base; all 6 loads per k-step are base+immediate.
// One-deep software pipeline: loads for k+1 issued before WMMAs of k.
template <int KX, bool XF32>
__device__ __forceinline__ void mm_accum(
    v8f acc[4][2],
    const float*  __restrict__ xf,   // fp32 A source (XF32)
    const __bf16* __restrict__ xb,   // bf16 A source (!XF32)
    const __bf16* __restrict__ W,    // [4H, KX] bf16 row-major
    int b0, int j0, int lm, int lk)
{
  const size_t aoff = (size_t)(b0 + lm) * KX + lk;   // A row tile 0
  const __bf16* __restrict__ bp = W + (size_t)(j0 + lm) * KX + lk;

  v16bf a0 = load_a_frag<XF32>(xf, xb, aoff);
  v16bf a1 = load_a_frag<XF32>(xf, xb, aoff + (size_t)16 * KX);
  v16bf vb[4];
#pragma unroll
  for (int g = 0; g < 4; ++g)
    vb[g] = *(const v16bf*)(bp + (size_t)g * HH * KX);

#pragma unroll
  for (int kb = 32; kb <= KX; kb += 32) {
    v16bf na0 = a0, na1 = a1, nb[4] = {vb[0], vb[1], vb[2], vb[3]};
    if (kb < KX) {   // constant-folded after unroll
      na0 = load_a_frag<XF32>(xf, xb, aoff + kb);
      na1 = load_a_frag<XF32>(xf, xb, aoff + (size_t)16 * KX + kb);
#pragma unroll
      for (int g = 0; g < 4; ++g)
        nb[g] = *(const v16bf*)(bp + (size_t)g * HH * KX + kb);
    }
#pragma unroll
    for (int g = 0; g < 4; ++g) {
      acc[g][0] = __builtin_amdgcn_wmma_f32_16x16x32_bf16(
          false, a0, false, vb[g], (short)0, acc[g][0], false, false);
      acc[g][1] = __builtin_amdgcn_wmma_f32_16x16x32_bf16(
          false, a1, false, vb[g], (short)0, acc[g][1], false, false);
    }
    a0 = na0; a1 = na1;
#pragma unroll
    for (int g = 0; g < 4; ++g) vb[g] = nb[g];
  }
}

// ---------------- fused LSTM cell ----------------
// One workgroup (4 waves, 128 threads) computes a 32(batch) x 64(hidden) tile.
// Each wave: 32 batch rows x 16 hidden columns, all 4 gates in-wave.
template <bool XF32, int KX>
__device__ __forceinline__ void lstm_cell(
    const float*  __restrict__ xf,     // fp32 input [B, KX]  (layer 0)
    const __bf16* __restrict__ xb,     // bf16 input [B, KX]  (layer 1)
    const __bf16* __restrict__ hprev,  // [B, H] bf16 previous hidden
    const __bf16* __restrict__ Wx,     // [4H, KX] bf16 row-major
    const __bf16* __restrict__ Wh,     // [4H, H]  bf16 row-major
    const float*  __restrict__ bias,   // [4H] fp32 (bih+bhh)
    float*        __restrict__ c,      // [B, H] fp32 cell state (in/out)
    __bf16*       __restrict__ hout,   // [B, H] bf16 new hidden
    float*        __restrict__ fout,   // optional [B, H] fp32 output (layer 1)
    int blk)                           // 0..31
{
  const int lane = threadIdx.x & 31;
  const int wid  = threadIdx.x >> 5;
  const int b0   = (blk & 3) * 32;            // batch row block (32 rows)
  const int j0   = (blk >> 2) * 64 + wid*16;  // hidden column block (per wave)
  const int lm   = lane & 15;
  const int lk   = (lane >> 4) * 16;

  v8f acc[4][2] = {};   // [gate i,f,g,o][row tile] f32 accumulators

  mm_accum<KX, XF32>(acc, xf, xb, Wx, b0, j0, lm, lk);      // x_t @ Wx^T
  mm_accum<HH, false>(acc, nullptr, hprev, Wh, b0, j0, lm, lk); // h @ Wh^T

  // ---- nonlinear epilogue ----
  const int j  = j0 + lm;
  const float bI = bias[0*HH + j];
  const float bF = bias[1*HH + j];
  const float bG = bias[2*HH + j];
  const float bO = bias[3*HH + j];
#pragma unroll
  for (int u = 0; u < 2; ++u) {
#pragma unroll
    for (int r = 0; r < 8; ++r) {
      const int m = b0 + u*16 + r + (lane >> 4) * 8;   // batch row (C layout)
      const size_t idx = (size_t)m * HH + j;
      const float si = fast_sigmoid(acc[0][u][r] + bI);
      const float sf = fast_sigmoid(acc[1][u][r] + bF);
      const float tg = fast_tanh   (acc[2][u][r] + bG);
      const float so = fast_sigmoid(acc[3][u][r] + bO);
      const float cn = sf * c[idx] + si * tg;
      c[idx] = cn;
      const float hn = so * fast_tanh(cn);
      hout[idx] = (__bf16)hn;
      if (fout) fout[idx] = hn;
    }
  }
}

// ---------------- pipelined step kernel ----------------
// blocks 0..31  : layer 0, time step t     (if t < T)
// blocks 32..63 : layer 1, time step t-1   (if t >= 1)
__global__ __launch_bounds__(128)
void lstm_step(const float*  __restrict__ x,
               const __bf16* __restrict__ w0x, const __bf16* __restrict__ w0h,
               const float*  __restrict__ b0,
               const __bf16* __restrict__ w1x, const __bf16* __restrict__ w1h,
               const float*  __restrict__ b1,
               __bf16* __restrict__ hA, __bf16* __restrict__ hB,
               float*  __restrict__ cA, float*  __restrict__ cB,
               float*  __restrict__ out, int t)
{
  const int blk = blockIdx.x;
  if (blk < 32) {
    if (t < TT) {
      lstm_cell<true, DD>(x + (size_t)t * BB * DD, nullptr,
                          hA + ((t + 1) & 1) * (BB * HH),   // read h[t-1]
                          w0x, w0h, b0, cA,
                          hA + (t & 1) * (BB * HH),         // write h[t]
                          nullptr, blk);
    }
  } else {
    const int s = t - 1;
    if (s >= 0) {
      lstm_cell<false, HH>(nullptr,
                           hA + (s & 1) * (BB * HH),        // input: layer0 h[s]
                           hB + ((s + 1) & 1) * (BB * HH),  // read own h[s-1]
                           w1x, w1h, b1, cB,
                           hB + (s & 1) * (BB * HH),        // write h[s]
                           out + (size_t)s * BB * HH, blk - 32);
    }
  }
}

// ---------------- host launcher ----------------
extern "C" void kernel_launch(void* const* d_in, const int* in_sizes, int n_in,
                              void* d_out, int out_size, void* d_ws, size_t ws_size,
                              hipStream_t stream) {
  const float* x    = (const float*)d_in[0];
  const float* Wih0 = (const float*)d_in[1];
  const float* Whh0 = (const float*)d_in[2];
  const float* bih0 = (const float*)d_in[3];
  const float* bhh0 = (const float*)d_in[4];
  const float* Wih1 = (const float*)d_in[5];
  const float* Whh1 = (const float*)d_in[6];
  const float* bih1 = (const float*)d_in[7];
  const float* bhh1 = (const float*)d_in[8];

  char* ws = (char*)d_ws;
  __bf16* w0x = (__bf16*)(ws + OFF_W0X);
  __bf16* w0h = (__bf16*)(ws + OFF_W0H);
  __bf16* w1x = (__bf16*)(ws + OFF_W1X);
  __bf16* w1h = (__bf16*)(ws + OFF_W1H);
  float*  b0  = (float*) (ws + OFF_B0);
  float*  b1  = (float*) (ws + OFF_B1);
  __bf16* hA  = (__bf16*)(ws + OFF_HA);
  __bf16* hB  = (__bf16*)(ws + OFF_HB);
  float*  cA  = (float*) (ws + OFF_CA);
  float*  cB  = (float*) (ws + OFF_CB);
  float*  out = (float*)d_out;

  // Prologue: weights -> bf16 (resident in L2 across all 512 steps),
  // bias pre-sum, zero h/c state. Deterministic, re-run every call.
  {
    int n;
    n = 2048 * 256;
    cvt_f32_to_bf16<<<(n + 255) / 256, 256, 0, stream>>>(Wih0, w0x, n);
    n = 2048 * 512;
    cvt_f32_to_bf16<<<(n + 255) / 256, 256, 0, stream>>>(Whh0, w0h, n);
    cvt_f32_to_bf16<<<(n + 255) / 256, 256, 0, stream>>>(Wih1, w1x, n);
    cvt_f32_to_bf16<<<(n + 255) / 256, 256, 0, stream>>>(Whh1, w1h, n);
    n = 2048;
    bias_sum<<<(n + 255) / 256, 256, 0, stream>>>(bih0, bhh0, b0, n);
    bias_sum<<<(n + 255) / 256, 256, 0, stream>>>(bih1, bhh1, b1, n);
    n = (int)(STATE_BYTES / 4);
    zero_u32<<<(n + 255) / 256, 256, 0, stream>>>((uint32_t*)(ws + OFF_HA), n);
  }

  // Pipelined recurrence: T+1 launches; layer0(t) || layer1(t-1) per launch.
  for (int t = 0; t <= TT; ++t) {
    lstm_step<<<64, 128, 0, stream>>>(x, w0x, w0h, b0, w1x, w1h, b1,
                                      hA, hB, cA, cB, out, t);
  }
}
